// RecurrentModel_43250320670956
// MI455X (gfx1250) — matrix operand order repaired and checked by reading, not compiled, not run on hardware
//
#include <hip/hip_runtime.h>
#include <cstddef>

// ---------------- model constants ----------------
#define DM     768          // d_model
#define DI     1536         // d_inner
#define DS     16           // d_state
#define DTR    48           // dt_rank
#define RR     8            // lora rank
#define SEQN   1024
#define VOCABN 50432
#define NBASE  6
#define NUP    6
#define NLOOPS 5
#define LORA_SC 2.0f
#define EPSV   1e-5f
#define NEGV   (-1e9f)
#define TILE_K 64

typedef __attribute__((ext_vector_type(16))) _Float16 v16h;
typedef __attribute__((ext_vector_type(8)))  float    v8f;

__device__ __forceinline__ float silu_f(float v) {
    return v / (1.0f + __expf(-v));
}

// B fragment: column nrow, K run [kb, kb+16). FULL -> no guards at all.
template<bool FULL>
__device__ __forceinline__ v16h load_bfrag(const float* __restrict__ W,
                                           int nrow, int kb, int N, int K, int ldw)
{
    v16h bf;
    if (FULL || nrow < N) {
        const float* wp = W + (size_t)nrow * ldw + kb;
        #pragma unroll
        for (int q = 0; q < 4; ++q) {
            float4 w4;
            if (FULL) {
                w4 = *reinterpret_cast<const float4*>(wp + q * 4);
            } else {
                w4 = make_float4(0.f, 0.f, 0.f, 0.f);
                if (kb + q * 4 < K)     // K % 4 == 0 -> full float4 in range
                    w4 = *reinterpret_cast<const float4*>(wp + q * 4);
            }
            bf[4 * q + 0] = (_Float16)w4.x;
            bf[4 * q + 1] = (_Float16)w4.y;
            bf[4 * q + 2] = (_Float16)w4.z;
            bf[4 * q + 3] = (_Float16)w4.w;
        }
    } else {
        #pragma unroll
        for (int j = 0; j < 16; ++j) bf[j] = (_Float16)0.0f;
    }
    return bf;
}

// =====================================================================
// Generic WMMA GEMM:  C[M,N] = beta*C + alpha * (A[M,K] @ W[N,K]^T) + bias[N]
// block = 256 threads (8 waves); each wave computes TWO adjacent 16x16 tiles
// (A fragment reused across both WMMAs); block spans 16 rows x 256 cols.
// 16x64 A tile staged via LDS (one b128 load+store per thread), 2 k-substeps
// x 2 n-tiles = 4 v_wmma_f32_16x16x32_f16 per staged tile.
// FULL=true: K % 64 == 0 && N % 256 == 0 -> zero guards (hot GEMMs).
// FULL=false: float4-granular K guards + per-row N guards (ragged GEMMs).
// Requires M % 16 == 0, K % 4 == 0, lda/ldw % 4 == 0 (true for all calls).
// =====================================================================
template<bool FULL>
__global__ __launch_bounds__(256)
void wmma_gemm_kernel(const float* __restrict__ A, const float* __restrict__ W,
                      const float* __restrict__ bias, float* __restrict__ C,
                      int M, int N, int K, int lda, int ldw, int ldc,
                      float alpha, int beta)
{
    __shared__ float4 As4[16 * 16];             // 16 rows x 64 floats = 4 KB
    const int tid  = threadIdx.x;
    const int wave = tid >> 5;
    const int lane = tid & 31;
    const int m0   = blockIdx.x * 16;
    const int n0   = blockIdx.y * 256 + wave * 32;   // wave covers [n0, n0+32)

    const int lm = lane & 15;
    const int hi = lane >> 4;                   // 0: lanes 0-15, 1: lanes 16-31
    v8f acc0 = {}, acc1 = {};

    // this thread's A-stage slot: row = tid>>4, float4 col = tid&15
    const int sr  = tid >> 4;
    const int sc4 = (tid & 15) * 4;

    const int ksteps = (K + TILE_K - 1) / TILE_K;
    for (int ks = 0; ks < ksteps; ++ks) {
        const int k0 = ks * TILE_K;

        // ---- stage 16x64 A tile: one float4 per thread ----
        {
            const int k = k0 + sc4;
            float4 v;
            if (FULL) {
                v = *reinterpret_cast<const float4*>(A + (size_t)(m0 + sr) * lda + k);
            } else {
                v = make_float4(0.f, 0.f, 0.f, 0.f);
                if (k < K)              // K % 4 == 0 -> full float4 in range
                    v = *reinterpret_cast<const float4*>(A + (size_t)(m0 + sr) * lda + k);
            }
            As4[tid] = v;
        }
        __syncthreads();

        #pragma unroll
        for (int s = 0; s < 2; ++s) {
            // A fragment (contiguous k order: k = 16*hi + element), float2 LDS reads
            const float2* arow = reinterpret_cast<const float2*>(&As4[lm * 16]);
            v16h af;
            #pragma unroll
            for (int v = 0; v < 8; ++v) {
                const int p2 = 16 * s + (hi ? 8 : 0) + v;
                const float2 e = arow[p2];
                af[2 * v]     = (_Float16)e.x;
                af[2 * v + 1] = (_Float16)e.y;
            }

            const int kb = k0 + 32 * s + (hi ? 16 : 0);
            const v16h bf0 = load_bfrag<FULL>(W, n0 + lm,      kb, N, K, ldw);
            const v16h bf1 = load_bfrag<FULL>(W, n0 + 16 + lm, kb, N, K, ldw);

            // uniform across the wave: EXEC all-ones requirement satisfied
            acc0 = __builtin_amdgcn_wmma_f32_16x16x32_f16(false, af, false, bf0,
                                                          (short)0, acc0, false, false);
            acc1 = __builtin_amdgcn_wmma_f32_16x16x32_f16(false, af, false, bf1,
                                                          (short)0, acc1, false, false);
        }
        __syncthreads();
    }

    // epilogue: VGPR r -> row (r or r+8), col = ncol
    auto store_tile = [&](const v8f& acc, int ncol) {
        if (FULL || ncol < N) {
            const float bv = bias ? bias[ncol] : 0.0f;
            #pragma unroll
            for (int r = 0; r < 8; ++r) {
                const int row = m0 + (hi ? (r + 8) : r);
                const size_t idx = (size_t)row * ldc + ncol;
                float v = alpha * acc[r] + bv;
                if (beta) v += C[idx];
                C[idx] = v;
            }
        }
    };
    store_tile(acc0, n0 + lm);
    store_tile(acc1, n0 + 16 + lm);
}

// =====================================================================
// causal depthwise conv1d (width 4) + bias + silu
// xz layout [L, 2*DI]; channel c reads column c of the first DI half
// =====================================================================
__global__ void conv_silu_kernel(const float* __restrict__ xz,
                                 const float* __restrict__ cw,
                                 const float* __restrict__ cb,
                                 float* __restrict__ xi, int L)
{
    const int idx = blockIdx.x * blockDim.x + threadIdx.x;
    if (idx >= L * DI) return;
    const int c = idx % DI;
    const int t = idx / DI;
    float acc = cb[c];
    #pragma unroll
    for (int j = 0; j < 4; ++j) {
        const int tt = t - 3 + j;
        if (tt >= 0) acc += cw[c * 4 + j] * xz[(size_t)tt * (2 * DI) + c];
    }
    xi[idx] = silu_f(acc);
}

__global__ void softplus_kernel(float* __restrict__ d, int n)
{
    const int i = blockIdx.x * blockDim.x + threadIdx.x;
    if (i >= n) return;
    const float v = d[i];
    d[i] = (v > 20.0f) ? v : log1pf(__expf(v));
}

// =====================================================================
// selective scan: one thread per channel, sequential over L,
// B/C rows staged in LDS in 32-step chunks, silu(z) gating fused.
// grid = DI/256 blocks of 256 threads.
// =====================================================================
#define TCHUNK 32
__global__ __launch_bounds__(256)
void ssm_scan_kernel(const float* __restrict__ xi, const float* __restrict__ delta,
                     const float* __restrict__ dbc, const float* __restrict__ Alog,
                     const float* __restrict__ Dp, const float* __restrict__ xz,
                     float* __restrict__ sy, int L)
{
    __shared__ float Bs[TCHUNK][DS];            // 2 KB
    __shared__ float Cs[TCHUNK][DS];            // 2 KB
    const int d = blockIdx.x * blockDim.x + threadIdx.x;   // channel, < DI exactly

    float Areg[DS], h[DS];
    #pragma unroll
    for (int s = 0; s < DS; ++s) {
        Areg[s] = -__expf(Alog[(size_t)d * DS + s]);
        h[s] = 0.0f;
    }
    const float Dd = Dp[d];

    for (int t0 = 0; t0 < L; t0 += TCHUNK) {
        __syncthreads();
        for (int i = threadIdx.x; i < TCHUNK * DS; i += 256) {
            const int tt = i >> 4;
            const int s  = i & 15;
            Bs[tt][s] = dbc[(size_t)(t0 + tt) * 80 + DTR + s];
            Cs[tt][s] = dbc[(size_t)(t0 + tt) * 80 + DTR + DS + s];
        }
        __syncthreads();
        for (int tt = 0; tt < TCHUNK; ++tt) {
            const int t = t0 + tt;
            const float dl = delta[(size_t)t * DI + d];
            const float u  = xi[(size_t)t * DI + d];
            const float du = dl * u;
            float y = 0.0f;
            #pragma unroll
            for (int s = 0; s < DS; ++s) {
                h[s] = h[s] * __expf(dl * Areg[s]) + du * Bs[tt][s];
                y += h[s] * Cs[tt][s];
            }
            const float zv = xz[(size_t)t * (2 * DI) + DI + d];
            sy[(size_t)t * DI + d] = (y + u * Dd) * silu_f(zv);
        }
    }
}

// =====================================================================
// fused residual + RMSNorm. one block per token, 256 threads x 3 elems.
// mode 0: res += x (stored);  out = norm(res) * w      (layer prelude)
// mode 1: out = norm(x) * w                            (loop norm, in-place ok)
// mode 2: out = norm(x + res) * w  (res untouched)     (final norm)
// =====================================================================
__global__ __launch_bounds__(256)
void rmsnorm_kernel(const float* __restrict__ x, float* __restrict__ res,
                    const float* __restrict__ w, float* __restrict__ out, int mode)
{
    __shared__ float red[8];
    __shared__ float scale_s;
    const int t   = blockIdx.x;
    const int tid = threadIdx.x;
    float v[3];
    float ssq = 0.0f;
    #pragma unroll
    for (int j = 0; j < 3; ++j) {
        const int i = tid + j * 256;
        float b = x[(size_t)t * DM + i];
        if (mode == 0) { b += res[(size_t)t * DM + i]; res[(size_t)t * DM + i] = b; }
        else if (mode == 2) { b += res[(size_t)t * DM + i]; }
        v[j] = b;
        ssq += b * b;
    }
    #pragma unroll
    for (int o = 16; o > 0; o >>= 1) ssq += __shfl_down(ssq, o, 32);  // wave32
    if ((tid & 31) == 0) red[tid >> 5] = ssq;
    __syncthreads();
    if (tid == 0) {
        float s = 0.0f;
        #pragma unroll
        for (int i = 0; i < 8; ++i) s += red[i];
        scale_s = rsqrtf(s / (float)DM + EPSV);
    }
    __syncthreads();
    const float sc = scale_s;
    #pragma unroll
    for (int j = 0; j < 3; ++j) {
        const int i = tid + j * 256;
        out[(size_t)t * DM + i] = v[j] * sc * w[i];
    }
}

__global__ void embed_kernel(const float* __restrict__ emb, const int* __restrict__ ids,
                             float* __restrict__ x, float* __restrict__ res)
{
    const int t   = blockIdx.x;
    const int tid = threadIdx.x;
    const int tok = ids[t];
    #pragma unroll
    for (int j = 0; j < 3; ++j) {
        const int i = tid + j * 256;
        x[(size_t)t * DM + i]   = emb[(size_t)tok * DM + i];
        res[(size_t)t * DM + i] = 0.0f;
    }
}

__global__ void add_row_kernel(float* __restrict__ x, const float* __restrict__ rowv)
{
    const int t   = blockIdx.x;
    const int tid = threadIdx.x;
    #pragma unroll
    for (int j = 0; j < 3; ++j) {
        const int i = tid + j * 256;
        x[(size_t)t * DM + i] += rowv[i];
    }
}

__global__ void mask_init_kernel(float* __restrict__ mask, int n)
{
    const int i = blockIdx.x * blockDim.x + threadIdx.x;
    if (i < n) mask[i] = NEGV;
}

__global__ void mask_scatter_kernel(const int* __restrict__ ids, float* __restrict__ mask, int L)
{
    const int t = blockIdx.x * blockDim.x + threadIdx.x;
    if (t < L) mask[ids[t]] = 0.0f;
}

// one wave per vocab row; xf (768) cached in LDS; float4 loads; wave32 shfl
__global__ __launch_bounds__(256)
void lmhead_kernel(const float* __restrict__ xf, const float* __restrict__ W,
                   const float* __restrict__ mask, float* __restrict__ out)
{
    __shared__ float4 xs4[DM / 4];              // 192 float4s
    const int tid = threadIdx.x;
    if (tid < DM / 4) xs4[tid] = reinterpret_cast<const float4*>(xf)[tid];
    __syncthreads();
    const int wave = tid >> 5;
    const int lane = tid & 31;
    const int v = blockIdx.x * 8 + wave;
    const float4* wp4 = reinterpret_cast<const float4*>(W + (size_t)v * DM);
    float s = 0.0f;
    #pragma unroll
    for (int k = 0; k < DM / 128; ++k) {        // 6 float4s per lane
        const float4 a = xs4[lane + k * 32];
        const float4 b = wp4[lane + k * 32];
        s += a.x * b.x + a.y * b.y + a.z * b.z + a.w * b.w;
    }
    #pragma unroll
    for (int o = 16; o > 0; o >>= 1) s += __shfl_down(s, o, 32);
    if (lane == 0) out[v] = s + mask[v];
}

// =====================================================================
// host orchestration
// =====================================================================
extern "C" void kernel_launch(void* const* d_in, const int* in_sizes, int n_in,
                              void* d_out, int out_size, void* d_ws, size_t ws_size,
                              hipStream_t stream)
{
    // ---- inputs in setup_inputs() dict order ----
    const float* emb_w    = (const float*)d_in[0];
    const float* lm_w     = (const float*)d_in[1];
    const float* normf_w  = (const float*)d_in[2];
    const float* loopn_w  = (const float*)d_in[3];
    const float* step_emb = (const float*)d_in[4];
    const float* b_in = (const float*)d_in[5];
    const float* b_cw = (const float*)d_in[6];
    const float* b_cb = (const float*)d_in[7];
    const float* b_xw = (const float*)d_in[8];
    const float* b_dw = (const float*)d_in[9];
    const float* b_db = (const float*)d_in[10];
    const float* b_Al = (const float*)d_in[11];
    const float* b_D  = (const float*)d_in[12];
    const float* b_ow = (const float*)d_in[13];
    const float* b_nw = (const float*)d_in[14];
    const float* u_in = (const float*)d_in[15];
    const float* u_cw = (const float*)d_in[16];
    const float* u_cb = (const float*)d_in[17];
    const float* u_xw = (const float*)d_in[18];
    const float* u_dw = (const float*)d_in[19];
    const float* u_db = (const float*)d_in[20];
    const float* u_Al = (const float*)d_in[21];
    const float* u_D  = (const float*)d_in[22];
    const float* u_ow = (const float*)d_in[23];
    const float* u_nw = (const float*)d_in[24];
    const float* u_iA = (const float*)d_in[25];
    const float* u_iB = (const float*)d_in[26];
    const float* u_oA = (const float*)d_in[27];
    const float* u_oB = (const float*)d_in[28];
    const float* c_in = (const float*)d_in[29];
    const float* c_cw = (const float*)d_in[30];
    const float* c_cb = (const float*)d_in[31];
    const float* c_xw = (const float*)d_in[32];
    const float* c_dw = (const float*)d_in[33];
    const float* c_db = (const float*)d_in[34];
    const float* c_Al = (const float*)d_in[35];
    const float* c_D  = (const float*)d_in[36];
    const float* c_ow = (const float*)d_in[37];
    const int*   ids  = (const int*)d_in[38];
    float* out = (float*)d_out;

    // ---- workspace carving ----
    float* p    = (float*)d_ws;
    float* x    = p; p += (size_t)SEQN * DM;
    float* res  = p; p += (size_t)SEQN * DM;
    float* h    = p; p += (size_t)SEQN * DM;
    float* xzb  = p; p += (size_t)SEQN * 2 * DI;
    float* xib  = p; p += (size_t)SEQN * DI;
    float* dbcb = p; p += (size_t)SEQN * 80;
    float* dlt  = p; p += (size_t)SEQN * DI;
    float* syb  = p; p += (size_t)SEQN * DI;
    float* tmp  = p; p += (size_t)SEQN * 16;
    float* mask = p; p += (size_t)VOCABN;
    float* xf   = p; p += (size_t)DM;

    auto gemm = [&](const float* A, const float* W, const float* bias, float* C,
                    int M, int N, int K, int lda, int ldw, int ldc,
                    float alpha, int beta) {
        dim3 grid(M / 16, (N + 255) / 256);
        const bool full = (K % TILE_K == 0) && (N % 256 == 0);
        if (full)
            wmma_gemm_kernel<true><<<grid, 256, 0, stream>>>(A, W, bias, C, M, N, K,
                                                             lda, ldw, ldc, alpha, beta);
        else
            wmma_gemm_kernel<false><<<grid, 256, 0, stream>>>(A, W, bias, C, M, N, K,
                                                              lda, ldw, ldc, alpha, beta);
    };

    // full mixer; LoRA factors applied on the activation side; out-proj writes
    // (beta=0) or accumulates (beta=1) into xdst.
    auto mixer = [&](const float* hin,
                     const float* in_w, const float* cw, const float* cb,
                     const float* xw, const float* dw, const float* db,
                     const float* Al, const float* Dp, const float* ow,
                     const float* iA, const float* iB,
                     const float* oA, const float* oB,
                     float* xdst, int beta_out) {
        gemm(hin, in_w, nullptr, xzb, SEQN, 2 * DI, DM, DM, DM, 2 * DI, 1.0f, 0);
        if (iA) {
            gemm(hin, iA, nullptr, tmp, SEQN, RR, DM, DM, DM, RR, 1.0f, 0);
            gemm(tmp, iB, nullptr, xzb, SEQN, 2 * DI, RR, RR, RR, 2 * DI, LORA_SC, 1);
        }
        conv_silu_kernel<<<(SEQN * DI + 255) / 256, 256, 0, stream>>>(xzb, cw, cb, xib, SEQN);
        gemm(xib, xw, nullptr, dbcb, SEQN, DTR + 2 * DS, DI, DI, DI, 80, 1.0f, 0);
        gemm(dbcb, dw, db, dlt, SEQN, DI, DTR, 80, DTR, DI, 1.0f, 0);
        softplus_kernel<<<(SEQN * DI + 255) / 256, 256, 0, stream>>>(dlt, SEQN * DI);
        ssm_scan_kernel<<<DI / 256, 256, 0, stream>>>(xib, dlt, dbcb, Al, Dp, xzb, syb, SEQN);
        gemm(syb, ow, nullptr, xdst, SEQN, DM, DI, DI, DI, DM, 1.0f, beta_out);
        if (oA) {
            gemm(syb, oA, nullptr, tmp, SEQN, RR, DI, DI, DI, RR, 1.0f, 0);
            gemm(tmp, oB, nullptr, xdst, SEQN, DM, RR, RR, RR, DM, LORA_SC, 1);
        }
    };

    // per-layer strides
    const size_t S_IN = (size_t)2 * DI * DM, S_CW = (size_t)DI * 4, S_V = DI;
    const size_t S_XW = (size_t)80 * DI,     S_DW = (size_t)DI * DTR;
    const size_t S_AL = (size_t)DI * DS,     S_OW = (size_t)DM * DI, S_N = DM;
    const size_t S_IA = (size_t)RR * DM,     S_IB = (size_t)2 * DI * RR;
    const size_t S_OA = (size_t)RR * DI,     S_OB = (size_t)DM * RR;

    auto run_up_stack = [&]() {
        for (int j = 0; j < NUP; ++j) {
            rmsnorm_kernel<<<SEQN, 256, 0, stream>>>(x, res, u_nw + j * S_N, h, 0);
            mixer(h, u_in + j * S_IN, u_cw + j * S_CW, u_cb + j * S_V,
                  u_xw + j * S_XW, u_dw + j * S_DW, u_db + j * S_V,
                  u_Al + j * S_AL, u_D + j * S_V, u_ow + j * S_OW,
                  u_iA + j * S_IA, u_iB + j * S_IB,
                  u_oA + j * S_OA, u_oB + j * S_OB, x, 0);
        }
    };

    // ---- forward ----
    embed_kernel<<<SEQN, 256, 0, stream>>>(emb_w, ids, x, res);

    for (int j = 0; j < NBASE; ++j) {
        rmsnorm_kernel<<<SEQN, 256, 0, stream>>>(x, res, b_nw + j * S_N, h, 0);
        mixer(h, b_in + j * S_IN, b_cw + j * S_CW, b_cb + j * S_V,
              b_xw + j * S_XW, b_dw + j * S_DW, b_db + j * S_V,
              b_Al + j * S_AL, b_D + j * S_V, b_ow + j * S_OW,
              nullptr, nullptr, nullptr, nullptr, x, 0);
    }

    run_up_stack();

    mask_init_kernel<<<(VOCABN + 255) / 256, 256, 0, stream>>>(mask, VOCABN);
    mask_scatter_kernel<<<(SEQN + 255) / 256, 256, 0, stream>>>(ids, mask, SEQN);

    for (int i = 0; i < NLOOPS; ++i) {
        add_row_kernel<<<SEQN, 256, 0, stream>>>(x, step_emb + (size_t)i * DM);
        run_up_stack();
        // x = x + mixer(x, core): fold the residual into the out-proj (beta=1)
        mixer(x, c_in, c_cw, c_cb, c_xw, c_dw, c_db, c_Al, c_D, c_ow,
              nullptr, nullptr, nullptr, nullptr, x, 1);
        rmsnorm_kernel<<<SEQN, 256, 0, stream>>>(x, nullptr, loopn_w, x, 1);
        // final norm needed only for the last token
        rmsnorm_kernel<<<1, 256, 0, stream>>>(x + (size_t)(SEQN - 1) * DM,
                                              res + (size_t)(SEQN - 1) * DM,
                                              normf_w, xf, 2);
        lmhead_kernel<<<VOCABN / 8, 256, 0, stream>>>(xf, lm_w, mask,
                                                      out + (size_t)i * VOCABN);
    }
}
